// Block_75840532513146
// MI455X (gfx1250) — matrix operand order repaired and checked by reading, not compiled, not run on hardware
//
#include <hip/hip_runtime.h>

// ---------------------------------------------------------------- types
typedef __attribute__((ext_vector_type(16))) __bf16 v16bf;
typedef __attribute__((ext_vector_type(8)))  float  v8f;

union FragU { v16bf v; unsigned u[8]; };

__device__ __forceinline__ unsigned short f2bf(float f) {
  unsigned u = __float_as_uint(f);
  u += 0x7fffu + ((u >> 16) & 1u);          // round-to-nearest-even
  return (unsigned short)(u >> 16);
}

// CDNA5 async DMA: global -> LDS, 16B per lane, tracked by ASYNCcnt.
// GVS mode: mem_addr = SGPR base + VGPR byte offset.
__device__ __forceinline__ void async_b128(unsigned lds_off, const void* sbase,
                                           unsigned byte_off) {
  asm volatile("global_load_async_to_lds_b128 %0, %1, %2"
               :: "v"(lds_off), "v"(byte_off), "s"(sbase)
               : "memory");
}
__device__ __forceinline__ void wait_async0() {
  asm volatile("s_wait_asynccnt 0x0" ::: "memory");
}
__device__ __forceinline__ unsigned lds_off_of(const void* p) {
  return (unsigned)(size_t)p;               // low 32 bits of generic = LDS offset
}

// A-fragment (16x32 bf16, M x K): lane = M (lane&15); lanes<16: v0..3 K=0..7,
// v4..7 K=16..23 ; lanes>=16: v0..3 K=8..15, v4..7 K=24..31  (ISA 7.12.2)
__device__ __forceinline__ v16bf load_afrag(const unsigned short* base,
                                            int m0, int k0, int ld) {
  int lane = threadIdx.x & 31;
  int m = m0 + (lane & 15);
  int hf = lane >> 4;
  const unsigned short* row = base + m * ld + k0 + hf * 8;
  FragU r;
#pragma unroll
  for (int j = 0; j < 4; ++j) {
    r.u[j]     = *(const unsigned*)(row + 2 * j);
    r.u[j + 4] = *(const unsigned*)(row + 16 + 2 * j);
  }
  return r.v;
}

// B-fragment (32x16 bf16, K x N), stored as [N][K]: lane = N (lane&15);
// lanes<16: v0..3 K=0..7, v4..7 K=16..23 ; lanes>=16: v0..3 K=8..15, v4..7 K=24..31
__device__ __forceinline__ v16bf load_bfrag(const unsigned short* base,
                                            int n0, int k0, int ld) {
  int lane = threadIdx.x & 31;
  int n = n0 + (lane & 15);
  int hf = lane >> 4;
  const unsigned short* row = base + n * ld + k0 + hf * 8;
  FragU r;
#pragma unroll
  for (int j = 0; j < 4; ++j) {
    r.u[j]     = *(const unsigned*)(row + 2 * j);
    r.u[j + 4] = *(const unsigned*)(row + 16 + 2 * j);
  }
  return r.v;
}

__device__ __forceinline__ v8f wmma_bf16(v16bf a, v16bf b, v8f c) {
  return __builtin_amdgcn_wmma_f32_16x16x32_bf16(false, a, false, b,
                                                 (short)0, c, false, false);
}

__device__ __forceinline__ v8f vzero8() {
  v8f z;
#pragma unroll
  for (int i = 0; i < 8; ++i) z[i] = 0.f;
  return z;
}

// ---------------------------------------------------------------- weight pack
// wq/wk/wv [H,C,D] fp32 -> Wt [3C][C] bf16 (row = output col, col = input c)
__global__ __launch_bounds__(256)
void pack_qkv_kernel(const float* __restrict__ wq, const float* __restrict__ wk,
                     const float* __restrict__ wv, unsigned short* __restrict__ out) {
  int idx = blockIdx.x * 256 + threadIdx.x;          // over 3C*C
  int n = idx >> 10;                                  // output col (0..3071)
  int k = idx & 1023;                                 // input channel
  int which = n >> 10;
  int rem = n & 1023;
  int h = rem >> 6, d = rem & 63;
  const float* src = (which == 0) ? wq : (which == 1) ? wk : wv;
  out[idx] = f2bf(src[(h * 1024 + k) * 64 + d]);
}

// in [K][N] fp32 -> out [N][K] bf16
__global__ __launch_bounds__(256)
void transpose_pack_kernel(const float* __restrict__ in,
                           unsigned short* __restrict__ out, int K, int N) {
  int idx = blockIdx.x * 256 + threadIdx.x;          // over N*K
  int n = idx / K;
  int k = idx - n * K;
  out[idx] = f2bf(in[(size_t)k * N + n]);
}

// ---------------------------------------------------------------- layernorm
__device__ __forceinline__ float block_sum(float v, float* buf) {
#pragma unroll
  for (int m = 16; m >= 1; m >>= 1) v += __shfl_xor(v, m, 32);
  int lane = threadIdx.x & 31, wid = threadIdx.x >> 5;
  if (lane == 0) buf[wid] = v;
  __syncthreads();
  float t = buf[0];
#pragma unroll
  for (int i = 1; i < 8; ++i) t += buf[i];
  return t;
}

__global__ __launch_bounds__(256)
void layernorm_kernel(const float* __restrict__ x, const float* __restrict__ g,
                      const float* __restrict__ be,
                      unsigned short* __restrict__ obf, float* __restrict__ of) {
  __shared__ float sred[16];
  int row = blockIdx.x, tid = threadIdx.x;
  const float* xr = x + (size_t)row * 1024;
  float v[4]; float s = 0.f;
#pragma unroll
  for (int i = 0; i < 4; ++i) { v[i] = xr[tid + 256 * i]; s += v[i]; }
  s = block_sum(s, sred);
  float mean = s * (1.f / 1024.f);
  float q = 0.f;
#pragma unroll
  for (int i = 0; i < 4; ++i) { float d = v[i] - mean; q += d * d; }
  q = block_sum(q, sred + 8);
  float inv = rsqrtf(q * (1.f / 1024.f) + 1e-3f);
#pragma unroll
  for (int i = 0; i < 4; ++i) {
    int c = tid + 256 * i;
    float o = (v[i] - mean) * inv * g[c] + be[c];
    if (obf) obf[(size_t)row * 1024 + c] = f2bf(o);
    if (of)  of[(size_t)row * 1024 + c] = o;
  }
}

// out = x1 + LN(y + z)
__global__ __launch_bounds__(256)
void final_ln_kernel(const float* __restrict__ x1, const float* __restrict__ y,
                     const float* __restrict__ z, const float* __restrict__ g,
                     const float* __restrict__ be, float* __restrict__ out) {
  __shared__ float sred[16];
  int row = blockIdx.x, tid = threadIdx.x;
  size_t base = (size_t)row * 1024;
  float v[4]; float s = 0.f;
#pragma unroll
  for (int i = 0; i < 4; ++i) {
    int c = tid + 256 * i;
    v[i] = y[base + c] + z[base + c];
    s += v[i];
  }
  s = block_sum(s, sred);
  float mean = s * (1.f / 1024.f);
  float q = 0.f;
#pragma unroll
  for (int i = 0; i < 4; ++i) { float d = v[i] - mean; q += d * d; }
  q = block_sum(q, sred + 8);
  float inv = rsqrtf(q * (1.f / 1024.f) + 1e-3f);
#pragma unroll
  for (int i = 0; i < 4; ++i) {
    int c = tid + 256 * i;
    out[base + c] = x1[base + c] + ((v[i] - mean) * inv * g[c] + be[c]);
  }
}

// ---------------------------------------------------------------- GEMM
// C[M,N] = A[M,K](bf16 row-major) * B[N,K](bf16) + epilogue.
// BK=64, double-buffered LDS filled by async global->LDS DMA (ASYNCcnt).
#define BM 128
#define BN 128
#define BKK 64

template <bool BIAS, bool RELU, bool RES, bool OBF>
__global__ __launch_bounds__(256)
void gemm_bf16_kernel(const unsigned short* __restrict__ A,
                      const unsigned short* __restrict__ Bt,
                      const float* __restrict__ bias,
                      const float* __restrict__ res,
                      void* __restrict__ outp, int M, int N, int K) {
  __shared__ __align__(16) unsigned short As[2][BM * BKK];
  __shared__ __align__(16) unsigned short Bs[2][BN * BKK];
  int tid = threadIdx.x;
  int bn = blockIdx.x * BN;
  int bm = blockIdx.y * BM;
  int lane = tid & 31, wid = tid >> 5;
  int wm = (wid & 3) * 32;    // wave 32-row slab
  int wn = (wid >> 2) * 64;   // wave 64-col slab

  auto prefetch = [&](int pb, int k0) {
#pragma unroll
    for (int p = 0; p < 4; ++p) {
      int ch = p * 256 + tid;               // 1024 chunks of 8 bf16 per matrix
      int r = ch >> 3, c = (ch & 7) * 8;
      async_b128(lds_off_of(&As[pb][r * BKK + c]), A,
                 (unsigned)(((size_t)(bm + r) * K + k0 + c) * 2));
      async_b128(lds_off_of(&Bs[pb][r * BKK + c]), Bt,
                 (unsigned)(((size_t)(bn + r) * K + k0 + c) * 2));
    }
  };

  v8f acc[2][4];
#pragma unroll
  for (int i = 0; i < 2; ++i)
#pragma unroll
    for (int jx = 0; jx < 4; ++jx) acc[i][jx] = vzero8();

  prefetch(0, 0);
  int buf = 0;
  for (int k0 = 0; k0 < K; k0 += BKK) {
    wait_async0();
    __syncthreads();                        // tile ready for all waves
    if (k0 + BKK < K) prefetch(buf ^ 1, k0 + BKK);
    const unsigned short* Ab = As[buf];
    const unsigned short* Bb = Bs[buf];
#pragma unroll
    for (int kk = 0; kk < BKK; kk += 32) {
      v16bf af[2], bfv[4];
#pragma unroll
      for (int mf = 0; mf < 2; ++mf) af[mf] = load_afrag(Ab, wm + mf * 16, kk, BKK);
#pragma unroll
      for (int nf = 0; nf < 4; ++nf) bfv[nf] = load_bfrag(Bb, wn + nf * 16, kk, BKK);
#pragma unroll
      for (int mf = 0; mf < 2; ++mf)
#pragma unroll
        for (int nf = 0; nf < 4; ++nf)
          acc[mf][nf] = wmma_bf16(af[mf], bfv[nf], acc[mf][nf]);
    }
    buf ^= 1;
  }

  int nloc = lane & 15, hf = lane >> 4;
#pragma unroll
  for (int mf = 0; mf < 2; ++mf)
#pragma unroll
    for (int nf = 0; nf < 4; ++nf)
#pragma unroll
      for (int r = 0; r < 8; ++r) {
        int row = bm + wm + mf * 16 + r + 8 * hf;
        int col = bn + wn + nf * 16 + nloc;
        float v = acc[mf][nf][r];
        if (BIAS) v += bias[col];
        if (RELU) v = fmaxf(v, 0.f);
        if (RES)  v += res[(size_t)row * N + col];
        if (OBF) ((unsigned short*)outp)[(size_t)row * N + col] = f2bf(v);
        else     ((float*)outp)[(size_t)row * N + col] = v;
      }
}

// ---------------------------------------------------------------- attention
// One block per (b,h). T=256, D=64. Flash-style streaming softmax, all in LDS.
// Only the diagonal 32-key block needs the causal mask.
#define VT_LD 264

__global__ __launch_bounds__(256)
void attention_kernel(const unsigned short* __restrict__ qkv,
                      unsigned short* __restrict__ attn_out) {
  __shared__ __align__(16) unsigned short Qs[256 * 64];
  __shared__ __align__(16) unsigned short Ks[256 * 64];
  __shared__ __align__(16) unsigned short Vt[64 * VT_LD];   // [d][t] transposed
  __shared__ __align__(16) unsigned short Ps[8 * 32 * 32];  // per-wave P tile

  int tid = threadIdx.x;
  int bh = blockIdx.x;
  int b = bh >> 4, h = bh & 15;
  const unsigned short* qrow = qkv + (size_t)(b * 256) * 3072;

  // Q,K tiles via async DMA; V loaded to VGPRs and transposed into LDS.
#pragma unroll
  for (int p = 0; p < 8; ++p) {
    int ch = p * 256 + tid;
    int t = ch >> 3, c8 = (ch & 7) * 8;
    async_b128(lds_off_of(&Qs[t * 64 + c8]), qrow,
               (unsigned)(((size_t)t * 3072 + h * 64 + c8) * 2));
    async_b128(lds_off_of(&Ks[t * 64 + c8]), qrow,
               (unsigned)(((size_t)t * 3072 + 1024 + h * 64 + c8) * 2));
    union { float4 f4; unsigned short us[8]; } vv;
    vv.f4 = *(const float4*)&qrow[(size_t)t * 3072 + 2048 + h * 64 + c8];
#pragma unroll
    for (int i = 0; i < 8; ++i) Vt[(c8 + i) * VT_LD + t] = vv.us[i];
  }
  wait_async0();
  __syncthreads();

  int lane = tid & 31, wid = tid >> 5;
  int nloc = lane & 15, hf = lane >> 4;
  int rm = wid * 32;                       // this wave's query rows [rm, rm+32)
  const float SCALE = 0.03125f;            // 1024^-0.5 (reference uses full C)

  // hoisted Q fragments: 2 m-frags x 2 k-steps
  v16bf aq[2][2];
#pragma unroll
  for (int mf = 0; mf < 2; ++mf)
#pragma unroll
    for (int kk = 0; kk < 2; ++kk)
      aq[mf][kk] = load_afrag(Qs, rm + mf * 16, kk * 32, 64);

  v8f o[2][4];
#pragma unroll
  for (int mf = 0; mf < 2; ++mf)
#pragma unroll
    for (int df = 0; df < 4; ++df) o[mf][df] = vzero8();
  float mrow[2][8], lsum[2][8];
#pragma unroll
  for (int mf = 0; mf < 2; ++mf)
#pragma unroll
    for (int r = 0; r < 8; ++r) { mrow[mf][r] = -1e30f; lsum[mf][r] = 0.f; }

  unsigned short* Pw = Ps + wid * 1024;

  auto process = [&](int j, bool masked) {
    // S = Q * K^T over this 32-key block
    v8f s[2][2];
#pragma unroll
    for (int mf = 0; mf < 2; ++mf)
#pragma unroll
      for (int nf = 0; nf < 2; ++nf) s[mf][nf] = vzero8();
#pragma unroll
    for (int kk = 0; kk < 2; ++kk) {
      v16bf bk0 = load_bfrag(Ks, j * 32, kk * 32, 64);
      v16bf bk1 = load_bfrag(Ks, j * 32 + 16, kk * 32, 64);
#pragma unroll
      for (int mf = 0; mf < 2; ++mf) {
        s[mf][0] = wmma_bf16(aq[mf][kk], bk0, s[mf][0]);
        s[mf][1] = wmma_bf16(aq[mf][kk], bk1, s[mf][1]);
      }
    }
    // streaming softmax update (mask only on the diagonal block)
#pragma unroll
    for (int mf = 0; mf < 2; ++mf) {
#pragma unroll
      for (int r = 0; r < 8; ++r) {
        float v0 = s[mf][0][r] * SCALE;
        float v1 = s[mf][1][r] * SCALE;
        if (masked) {
          int row = rm + mf * 16 + r + 8 * hf;
          int c0 = j * 32 + nloc;
          if (c0 > row) v0 = -1e30f;
          if (c0 + 16 > row) v1 = -1e30f;
        }
        float mx = fmaxf(v0, v1);
        mx = fmaxf(mx, __shfl_xor(mx, 1, 32));
        mx = fmaxf(mx, __shfl_xor(mx, 2, 32));
        mx = fmaxf(mx, __shfl_xor(mx, 4, 32));
        mx = fmaxf(mx, __shfl_xor(mx, 8, 32));
        float nm = fmaxf(mrow[mf][r], mx);
        float alpha = __expf(mrow[mf][r] - nm);
        float p0 = __expf(v0 - nm);
        float p1 = __expf(v1 - nm);
        float ps = p0 + p1;
        ps += __shfl_xor(ps, 1, 32);
        ps += __shfl_xor(ps, 2, 32);
        ps += __shfl_xor(ps, 4, 32);
        ps += __shfl_xor(ps, 8, 32);
        lsum[mf][r] = lsum[mf][r] * alpha + ps;
        mrow[mf][r] = nm;
#pragma unroll
        for (int df = 0; df < 4; ++df) o[mf][df][r] *= alpha;
        int ml = mf * 16 + r + 8 * hf;
        Pw[ml * 32 + nloc]      = f2bf(p0);
        Pw[ml * 32 + 16 + nloc] = f2bf(p1);
      }
    }
    asm volatile("s_wait_dscnt 0x0" ::: "memory");  // P stores -> frag loads
    // O += P * V
#pragma unroll
    for (int mf = 0; mf < 2; ++mf) {
      v16bf pa = load_afrag(Pw, mf * 16, 0, 32);
#pragma unroll
      for (int df = 0; df < 4; ++df) {
        v16bf vb = load_bfrag(Vt, df * 16, j * 32, VT_LD);
        o[mf][df] = wmma_bf16(pa, vb, o[mf][df]);
      }
    }
  };

  for (int j = 0; j < wid; ++j) process(j, false);  // fully-visible blocks
  process(wid, true);                               // diagonal (masked) block

  // normalize and store (bf16) into [B*T, C] at column h*64
  unsigned short* ao = attn_out + (size_t)(b * 256) * 1024 + h * 64;
#pragma unroll
  for (int mf = 0; mf < 2; ++mf)
#pragma unroll
    for (int r = 0; r < 8; ++r) {
      int t = rm + mf * 16 + r + 8 * hf;
      float rs = 1.0f / lsum[mf][r];
#pragma unroll
      for (int df = 0; df < 4; ++df)
        ao[(size_t)t * 1024 + df * 16 + nloc] = f2bf(o[mf][df][r] * rs);
    }
}

// ---------------------------------------------------------------- launch
extern "C" void kernel_launch(void* const* d_in, const int* in_sizes, int n_in,
                              void* d_out, int out_size, void* d_ws, size_t ws_size,
                              hipStream_t stream) {
  const float* x      = (const float*)d_in[0];
  const float* wq     = (const float*)d_in[1];
  const float* wk     = (const float*)d_in[2];
  const float* wv     = (const float*)d_in[3];
  const float* w_proj = (const float*)d_in[4];
  const float* b_proj = (const float*)d_in[5];
  const float* w1     = (const float*)d_in[6];
  const float* b1     = (const float*)d_in[7];
  const float* w2     = (const float*)d_in[8];
  const float* b2     = (const float*)d_in[9];
  const float* g1     = (const float*)d_in[10];
  const float* be1    = (const float*)d_in[11];
  const float* g2     = (const float*)d_in[12];
  const float* be2    = (const float*)d_in[13];
  const float* g3     = (const float*)d_in[14];
  const float* be3    = (const float*)d_in[15];

  const int M = 64 * 256;   // 16384 tokens
  const int C = 1024;

  char* wsb = (char*)d_ws;
  size_t off = 0;
  auto alloc = [&](size_t bytes) -> char* {
    char* p = wsb + off;
    off = (off + bytes + 255) & ~(size_t)255;
    return p;
  };
  unsigned short* h_bf    = (unsigned short*)alloc((size_t)M * C * 2);
  unsigned short* wqkv_t  = (unsigned short*)alloc((size_t)3 * C * C * 2);
  unsigned short* wproj_t = (unsigned short*)alloc((size_t)C * C * 2);
  unsigned short* w1_t    = (unsigned short*)alloc((size_t)4 * C * C * 2);
  unsigned short* w2_t    = (unsigned short*)alloc((size_t)4 * C * C * 2);
  unsigned short* qkv     = (unsigned short*)alloc((size_t)M * 3 * C * 2);
  unsigned short* attn    = (unsigned short*)alloc((size_t)M * C * 2);
  float*          x1      = (float*)alloc((size_t)M * C * 4);
  unsigned short* y_bf    = (unsigned short*)alloc((size_t)M * C * 2);
  float*          y_f     = (float*)alloc((size_t)M * C * 4);
  unsigned short* act     = (unsigned short*)alloc((size_t)M * 4 * C * 2);
  float*          z_f     = (float*)alloc((size_t)M * C * 4);
  (void)ws_size; (void)n_in; (void)in_sizes; (void)out_size;

  // weight packing (bf16, [N][K])
  pack_qkv_kernel<<<3 * C * C / 256, 256, 0, stream>>>(wq, wk, wv, wqkv_t);
  transpose_pack_kernel<<<C * C / 256, 256, 0, stream>>>(w_proj, wproj_t, C, C);
  transpose_pack_kernel<<<4 * C * C / 256, 256, 0, stream>>>(w1, w1_t, C, 4 * C);
  transpose_pack_kernel<<<4 * C * C / 256, 256, 0, stream>>>(w2, w2_t, 4 * C, C);

  // h = LN1(x) -> bf16
  layernorm_kernel<<<M, 256, 0, stream>>>(x, g1, be1, h_bf, nullptr);

  // qkv = h @ Wqkv  (no bias, bf16 out)
  gemm_bf16_kernel<false, false, false, true>
      <<<dim3(3 * C / BN, M / BM), 256, 0, stream>>>(
          h_bf, wqkv_t, nullptr, nullptr, qkv, M, 3 * C, C);

  // attention per (b,h)
  attention_kernel<<<64 * 16, 256, 0, stream>>>(qkv, attn);

  // x1 = x + attn @ Wproj + b_proj   (fp32 out)
  gemm_bf16_kernel<true, false, true, false>
      <<<dim3(C / BN, M / BM), 256, 0, stream>>>(
          attn, wproj_t, b_proj, x, x1, M, C, C);

  // y = LN2(x1) -> bf16 + fp32
  layernorm_kernel<<<M, 256, 0, stream>>>(x1, g2, be2, y_bf, y_f);

  // act = relu(y @ W1 + b1) -> bf16
  gemm_bf16_kernel<true, true, false, true>
      <<<dim3(4 * C / BN, M / BM), 256, 0, stream>>>(
          y_bf, w1_t, b1, nullptr, act, M, 4 * C, C);

  // z = act @ W2 + b2 -> fp32
  gemm_bf16_kernel<true, false, false, false>
      <<<dim3(C / BN, M / BM), 256, 0, stream>>>(
          act, w2_t, b2, nullptr, z_f, M, C, 4 * C);

  // out = x1 + LN3(y + z)
  final_ln_kernel<<<M, 256, 0, stream>>>(x1, y_f, z_f, g3, be3, (float*)d_out);
}
